// SoftArgMax_33148557591200
// MI455X (gfx1250) — matrix operand order repaired and checked by reading, not compiled, not run on hardware
//
#include <hip/hip_runtime.h>
#include <stdint.h>

#define S_DIM 1024
#define V_DIM 32768
#define M_DIM 384
#define BETA_F 100.0f

typedef __attribute__((ext_vector_type(16))) __bf16 v16bf;
typedef __attribute__((ext_vector_type(8)))  float  v8f;

// fp32 -> bf16, round-to-nearest-even
__device__ __forceinline__ unsigned short f2bf(float x) {
    unsigned u = __float_as_uint(x);
    unsigned r = u + 0x7FFFu + ((u >> 16) & 1u);
    return (unsigned short)(r >> 16);
}

// -----------------------------------------------------------------------------
// Kernel 1: row softmax over S=1024, emit bf16 A in WMMA A-fragment order.
// A-fragment (16-bit 16x32): lane l (m = l&15, kb = l<16 ? 0 : 16) holds
// slots j=0..15 -> K = kb+j (contiguous, matches B). One 1KB block
// per (m-tile, k-tile); lane's 16 bf16 are contiguous at lane*32 bytes.
// -----------------------------------------------------------------------------
__global__ __launch_bounds__(256) void softmax_to_bf16_frag(
        const float* __restrict__ inp, unsigned short* __restrict__ Aws) {
    __shared__ float red[256];
    const int row = blockIdx.x;           // 0..383
    const int t   = threadIdx.x;
    const float* rp = inp + (size_t)row * S_DIM;

    float x0 = rp[t], x1 = rp[t + 256], x2 = rp[t + 512], x3 = rp[t + 768];
    float mx = fmaxf(fmaxf(x0, x1), fmaxf(x2, x3));
    red[t] = mx; __syncthreads();
    for (int s = 128; s > 0; s >>= 1) {
        if (t < s) red[t] = fmaxf(red[t], red[t + s]);
        __syncthreads();
    }
    mx = red[0]; __syncthreads();

    float e0 = __expf(BETA_F * (x0 - mx));
    float e1 = __expf(BETA_F * (x1 - mx));
    float e2 = __expf(BETA_F * (x2 - mx));
    float e3 = __expf(BETA_F * (x3 - mx));
    red[t] = e0 + e1 + e2 + e3; __syncthreads();
    for (int s = 128; s > 0; s >>= 1) {
        if (t < s) red[t] += red[t + s];
        __syncthreads();
    }
    const float inv = 1.0f / red[0];

    const int mt = row >> 4, mrow = row & 15;
    unsigned short* base = Aws + (size_t)mt * 32 * 512;  // 32 k-tiles * 512 shorts
    float vals[4] = { e0 * inv, e1 * inv, e2 * inv, e3 * inv };
    int   ks[4]   = { t, t + 256, t + 512, t + 768 };
    #pragma unroll
    for (int i = 0; i < 4; ++i) {
        int k = ks[i];
        int kt = k >> 5, kk = k & 31;
        int lane = mrow + ((kk & 16) ? 16 : 0);
        int j    = kk & 15;
        base[(size_t)kt * 512 + lane * 16 + j] = f2bf(vals[i]);
    }
}

// -----------------------------------------------------------------------------
// Kernel 2: voxels fp32 [1024 x 32768] -> bf16 B-fragment blocks via LDS.
// B-fragment (16-bit 32x16): lane l (n = l&15, kb = l<16 ? 0 : 16) holds
// K = kb..kb+15 contiguous (VGPR v = K {2v, 2v+1}). One 1KB block per
// (n-tile, k-tile), block index = nt*32 + kt so the GEMM k-loop is contiguous.
// -----------------------------------------------------------------------------
__global__ __launch_bounds__(256) void voxels_to_bf16_frag(
        const float* __restrict__ vox, unsigned short* __restrict__ Bws) {
    __shared__ unsigned short lds[32 * 256];   // 16 KB
    const int t     = threadIdx.x;
    const int kt    = blockIdx.x >> 7;         // 0..31
    const int nblk  = blockIdx.x & 127;        // 0..127
    const int ncol0 = nblk * 256;

    // Phase 1: coalesced read of 32 rows x 256 cols, convert to bf16 in LDS
    {
        int k = t >> 3, seg = t & 7;
        const float4* src = reinterpret_cast<const float4*>(
            vox + (size_t)(kt * 32 + k) * V_DIM + ncol0 + seg * 32);
        unsigned short* dst = lds + k * 256 + seg * 32;
        #pragma unroll
        for (int q = 0; q < 8; ++q) {
            float4 f = src[q];
            dst[q * 4 + 0] = f2bf(f.x);
            dst[q * 4 + 1] = f2bf(f.y);
            dst[q * 4 + 2] = f2bf(f.z);
            dst[q * 4 + 3] = f2bf(f.w);
        }
    }
    __syncthreads();

    // Phase 2: emit 16 fragment blocks (1 KB each) in lane-register order
    {
        int tl  = t >> 4;                  // tile_local 0..15
        int sub = t & 15;                  // n-within-tile
        int nt  = nblk * 16 + tl;
        size_t tile_off = ((size_t)nt * 32 + kt) * 512;   // shorts
        int nl  = tl * 16 + sub;           // column in LDS region
        #pragma unroll
        for (int half = 0; half < 2; ++half) {
            int lane = sub + half * 16;
            int kb   = half * 16;
            unsigned p[8];
            #pragma unroll
            for (int i = 0; i < 8; ++i) {
                unsigned lo = lds[(kb + 2 * i)     * 256 + nl];
                unsigned hi = lds[(kb + 2 * i + 1) * 256 + nl];
                p[i] = lo | (hi << 16);
            }
            uint4* outp = reinterpret_cast<uint4*>(Bws + tile_off + lane * 16);
            outp[0] = make_uint4(p[0], p[1], p[2], p[3]);
            outp[1] = make_uint4(p[4], p[5], p[6], p[7]);
        }
    }
}

// -----------------------------------------------------------------------------
// Kernel 3: GEMM, explicit double-buffered software pipeline with scheduling
// barriers. Each wave: one 16-M tile x 4 16-N tiles, 32 k-steps of
// v_wmma_f32_16x16x32_bf16. Per iteration: [prefetch kt+1 fragments]
// sched_barrier [4 WMMAs on kt] sched_barrier. The barriers stop the backend
// from sinking the prefetch below the WMMAs, so every WMMA quad waits only
// for the PREVIOUS iteration's 10 b128 loads while the next 10 stay in
// flight. ~115 VGPRs: no MSB addressing, multiple waves/SIMD.
// -----------------------------------------------------------------------------
__global__ __launch_bounds__(256) void gemm_wmma_bf16(
        const unsigned short* __restrict__ Aws,
        const unsigned short* __restrict__ Bws,
        float* __restrict__ out) {
    const int mt     = blockIdx.x >> 6;       // 0..23
    const int nsuper = blockIdx.x & 63;       // 0..63
    const int wave   = threadIdx.x >> 5;      // 0..7
    const int lane   = threadIdx.x & 31;
    const int nt0    = (nsuper * 8 + wave) * 4;   // first of 4 n-tiles

    const unsigned short* Ab = Aws + (size_t)mt * 32 * 512 + lane * 16;
    const unsigned short* B0 = Bws + (size_t)(nt0 + 0) * 32 * 512 + lane * 16;
    const unsigned short* B1 = Bws + (size_t)(nt0 + 1) * 32 * 512 + lane * 16;
    const unsigned short* B2 = Bws + (size_t)(nt0 + 2) * 32 * 512 + lane * 16;
    const unsigned short* B3 = Bws + (size_t)(nt0 + 3) * 32 * 512 + lane * 16;

    v8f acc0 = {}, acc1 = {}, acc2 = {}, acc3 = {};

    v16bf a[2], b0[2], b1[2], b2[2], b3[2];
    a[0]  = *reinterpret_cast<const v16bf*>(Ab);
    b0[0] = *reinterpret_cast<const v16bf*>(B0);
    b1[0] = *reinterpret_cast<const v16bf*>(B1);
    b2[0] = *reinterpret_cast<const v16bf*>(B2);
    b3[0] = *reinterpret_cast<const v16bf*>(B3);

    #pragma unroll 2
    for (int kt = 0; kt < 32; ++kt) {
        const int cur = kt & 1;
        const int nxt = cur ^ 1;
        // prefetch next k-step (clamped at the end: one redundant reload of
        // tile 31, keeps the trip count even for the unroll-2 ping-pong)
        const size_t ko = (size_t)((kt < 31) ? (kt + 1) : 31) * 512;
        a[nxt]  = *reinterpret_cast<const v16bf*>(Ab + ko);
        b0[nxt] = *reinterpret_cast<const v16bf*>(B0 + ko);
        b1[nxt] = *reinterpret_cast<const v16bf*>(B1 + ko);
        b2[nxt] = *reinterpret_cast<const v16bf*>(B2 + ko);
        b3[nxt] = *reinterpret_cast<const v16bf*>(B3 + ko);
        // keep the prefetch above the WMMAs: nothing may cross
        __builtin_amdgcn_sched_barrier(0);
        // consume current buffers (loaded one full iteration ago)
        acc0 = __builtin_amdgcn_wmma_f32_16x16x32_bf16(false, a[cur], false, b0[cur], (short)0, acc0, false, false);
        acc1 = __builtin_amdgcn_wmma_f32_16x16x32_bf16(false, a[cur], false, b1[cur], (short)0, acc1, false, false);
        acc2 = __builtin_amdgcn_wmma_f32_16x16x32_bf16(false, a[cur], false, b2[cur], (short)0, acc2, false, false);
        acc3 = __builtin_amdgcn_wmma_f32_16x16x32_bf16(false, a[cur], false, b3[cur], (short)0, acc3, false, false);
        __builtin_amdgcn_sched_barrier(0);
    }

    // C/D layout: VGPR i -> M = i + 8*(lane>=16), N = lane&15
    const int ncol = lane & 15;
    const int mhi  = (lane >> 4) * 8;
    float* obase = out + (size_t)(mt * 16) * V_DIM;
    #pragma unroll
    for (int i = 0; i < 8; ++i) {
        size_t r = (size_t)(mhi + i) * V_DIM;
        obase[r + (size_t)(nt0 + 0) * 16 + ncol] = acc0[i];
        obase[r + (size_t)(nt0 + 1) * 16 + ncol] = acc1[i];
        obase[r + (size_t)(nt0 + 2) * 16 + ncol] = acc2[i];
        obase[r + (size_t)(nt0 + 3) * 16 + ncol] = acc3[i];
    }
}

extern "C" void kernel_launch(void* const* d_in, const int* in_sizes, int n_in,
                              void* d_out, int out_size, void* d_ws, size_t ws_size,
                              hipStream_t stream) {
    (void)in_sizes; (void)n_in; (void)out_size; (void)ws_size;
    const float* inp = (const float*)d_in[0];   // [384, 1024]
    const float* vox = (const float*)d_in[1];   // [1024, 32768]
    float* out = (float*)d_out;                 // [384, 32768]

    unsigned short* Aws = (unsigned short*)d_ws;                               // 768 KB
    unsigned short* Bws = (unsigned short*)((char*)d_ws + (size_t)(1u << 20)); // 64 MB

    softmax_to_bf16_frag<<<M_DIM, 256, 0, stream>>>(inp, Aws);
    voxels_to_bf16_frag<<<32 * 128, 256, 0, stream>>>(vox, Bws);
    gemm_wmma_bf16<<<24 * 64, 256, 0, stream>>>(Aws, Bws, out);
}
